// PIEGNN_3831110828535
// MI455X (gfx1250) — compile-verified, matrix operand order checked by reading, not run on hardware
//
#include <hip/hip_runtime.h>
#include <hip/hip_bf16.h>

typedef __attribute__((ext_vector_type(16))) _Float16 v16h;
typedef __attribute__((ext_vector_type(8)))  _Float16 v8h;
typedef __attribute__((ext_vector_type(8)))  float    v8f;
typedef __attribute__((ext_vector_type(4)))  unsigned uv4;
typedef __attribute__((ext_vector_type(8)))  int      iv8;
typedef __attribute__((ext_vector_type(4)))  int      iv4;

#define DEVINL __device__ __forceinline__

#if __has_builtin(__builtin_amdgcn_tensor_load_to_lds)
#define HAS_TDM 1
#else
#define HAS_TDM 0
#endif

namespace {
constexpr int kN  = 150000;   // nodes
constexpr int kS  = 10000;    // symbols
constexpr int kC  = 20000;    // clauses
constexpr int kM  = 200000;   // edges per type
constexpr int kMS = 150000;   // symbol-edge rows
constexpr int kNC = 200000;
constexpr int kCN = 200000;
constexpr int kB  = 64;       // graphs
}

// ---------------- monotonic float<->uint encoding for atomic max/min ----------
DEVINL unsigned fenc(float f) {
  unsigned b = __float_as_uint(f);
  return b ^ ((b >> 31) ? 0xFFFFFFFFu : 0x80000000u);
}
DEVINL float fdec(unsigned e) {
  unsigned b = e ^ ((e >> 31) ? 0x80000000u : 0xFFFFFFFFu);
  return __uint_as_float(b);
}

// ---------------- weight slab staging: global [128,K] f16 -> LDS 128x32 tile --
// TDM path: 2D D# descriptor, tile_dim = 32 x 128, dim0 stride = K elements.
// Wave 0 issues; completion fenced with s_wait_tensorcnt before the barrier.
DEVINL void stage_slab(const _Float16* __restrict__ Wg, int K, int k0,
                       _Float16* __restrict__ dst) {
#if HAS_TDM
  if (threadIdx.x < 32) {
    unsigned long long ga = (unsigned long long)(const void*)(Wg + (size_t)k0);
    unsigned lds = (unsigned)(unsigned long long)(uintptr_t)dst; // LDS byte offset
    uv4 g0;
    g0[0] = 1u;                                  // count=1, user descriptor
    g0[1] = lds;                                 // lds_addr
    g0[2] = (unsigned)ga;                        // global_addr[31:0]
    g0[3] = (unsigned)((ga >> 32) & 0x01FFFFFFu) | (2u << 30); // addr[56:32]|type=2
    iv8 g1;
    g1[0] = (int)(1u << 16);                     // data_size=1 (2 bytes)
    g1[1] = (int)(((unsigned)K & 0xFFFFu) << 16);        // tensor_dim0[15:0]
    g1[2] = (int)((((unsigned)K >> 16) & 0xFFFFu) | (128u << 16)); // dim0 hi | tensor_dim1 lo
    g1[3] = (int)(32u << 16);                    // tensor_dim1 hi=0 | tile_dim0=32
    g1[4] = 128;                                 // tile_dim1=128, tile_dim2=0
    g1[5] = K;                                   // tensor_dim0_stride[31:0]
    g1[6] = 0;
    g1[7] = 0;
    iv4 z4; z4[0] = 0; z4[1] = 0; z4[2] = 0; z4[3] = 0;
    iv8 z8;
#pragma unroll
    for (int i = 0; i < 8; ++i) z8[i] = 0;
    // 6-arg form (clang-23 / therock headers): (g0, g1, g2, g3, g4, cpol)
    __builtin_amdgcn_tensor_load_to_lds(g0, g1, z4, z4, z8, 0);
  }
#else
  // cooperative copy: 256 threads x 16 halfs = 128x32 tile
  const int e0   = threadIdx.x * 16;
  const int wcol = e0 >> 5;
  const int wkk  = e0 & 31;
  const _Float16* g = Wg + (size_t)wcol * K + k0 + wkk;
  if (k0 + 32 < K) __builtin_prefetch(g + 32, 0, 3);
  v8h x0 = *(const v8h*)g;
  v8h x1 = *(const v8h*)(g + 8);
  *(v8h*)(dst + e0)     = x0;
  *(v8h*)(dst + e0 + 8) = x1;
#endif
}

DEVINL void stage_wait() {
#if HAS_TDM
#if __has_builtin(__builtin_amdgcn_s_wait_tensorcnt)
  if (threadIdx.x < 32) __builtin_amdgcn_s_wait_tensorcnt(0);
#endif
#endif
}

// ---------------- WMMA GEMM core: 128 rows x 128 cols per block ---------------
// 8 waves; each wave: 16 rows x 128 cols (8 accumulators of v8f).
// W16 is [128, K] f16 row-major, K % 32 == 0; K-slabs double-buffered in LDS.
template <typename FetchA>
DEVINL void gemm_core(FetchA fetchA, const _Float16* __restrict__ Wg,
                      _Float16* ws0, _Float16* ws1, int K,
                      float* __restrict__ Y, int rows, int r0,
                      const float* __restrict__ bias,
                      const float* __restrict__ rowscale,
                      int accum, int relu) {
  const int lane = threadIdx.x & 31;
  const int hseg = lane >> 4;       // half-wave K-interleave select
  const int nlo  = lane & 15;       // output column within 16-wide tile

  v8f c[8];
#pragma unroll
  for (int t = 0; t < 8; ++t)
#pragma unroll
    for (int i = 0; i < 8; ++i) c[t][i] = 0.f;

  // prologue: stage slab 0
  stage_slab(Wg, K, 0, ws0);
  stage_wait();
  __syncthreads();

  _Float16* cur = ws0;
  _Float16* nxt = ws1;

  for (int k0 = 0; k0 < K; k0 += 32) {
    if (k0 + 32 < K) stage_slab(Wg, K, k0 + 32, nxt);   // overlap with compute

    v16h a = fetchA(k0);                 // per-lane A fragment (16 f16)

    v8h bl[8], bh[8];                    // batched LDS loads -> single wait
#pragma unroll
    for (int t = 0; t < 8; ++t) {
      const _Float16* wr = cur + (t * 16 + nlo) * 32 + hseg * 8;
      bl[t] = *(const v8h*)wr;           // K {0..7 | 8..15}
      bh[t] = *(const v8h*)(wr + 16);    // K {16..23 | 24..31}
    }
#pragma unroll
    for (int t = 0; t < 8; ++t) {
      v16h b;
#pragma unroll
      for (int i = 0; i < 8; ++i) { b[i] = bl[t][i]; b[8 + i] = bh[t][i]; }
      c[t] = __builtin_amdgcn_wmma_f32_16x16x32_f16(
          false, a, false, b, (short)0, c[t], false, false);
    }

    stage_wait();                        // next slab landed
    __syncthreads();                     // everyone done reading `cur`
    _Float16* tmp = cur; cur = nxt; nxt = tmp;
  }

  // epilogue: C/D layout -> m = r0 + i + hseg*8, n = t*16 + nlo
#pragma unroll
  for (int t = 0; t < 8; ++t) {
    const int n  = t * 16 + nlo;
    const float bv = bias ? bias[n] : 0.f;
#pragma unroll
    for (int i = 0; i < 8; ++i) {
      const int m = r0 + i + hseg * 8;
      if (m < rows) {
        float y = c[t][i] + bv;
        if (rowscale) y *= rowscale[m];
        const size_t off = (size_t)m * 128 + n;
        if (accum) y += Y[off];
        if (relu) y = fmaxf(y, 0.f);
        Y[off] = y;
      }
    }
  }
}

// ---- plain GEMM: Y[rows,128] (+)= act(A[rows,K](lda) @ W^T + bias) ----------
__global__ void k_gemm_plain(const float* __restrict__ A, int lda,
                             const _Float16* __restrict__ W16, int K,
                             const float* __restrict__ bias,
                             const float* __restrict__ rowscale,
                             float* __restrict__ Y, int rows,
                             int accum, int relu) {
  __shared__ _Float16 WS[2][128 * 32];
  const int lane = threadIdx.x & 31;
  const int hseg = lane >> 4;
  const int nlo  = lane & 15;
  const int r0   = blockIdx.x * 128 + (threadIdx.x >> 5) * 16;
  int r = r0 + nlo; if (r >= rows) r = rows - 1;
  const float* rowp = A + (size_t)r * lda;
  auto fetch = [&](int k0) {
    v8f lo = *(const v8f*)(rowp + k0 + hseg * 8);
    v8f hi = *(const v8f*)(rowp + k0 + 16 + hseg * 8);
    v16h a;
#pragma unroll
    for (int i = 0; i < 8; ++i) { a[i] = (_Float16)lo[i]; a[8 + i] = (_Float16)hi[i]; }
    return a;
  };
  gemm_core(fetch, W16, WS[0], WS[1], K, Y, rows, r0, bias, rowscale, accum, relu);
}

// ---- edge GEMM: A row m = [nodes[i0] | nodes[i1] | sgn*symbols[si]], K=3*din -
__global__ void k_gemm_edge(const float* __restrict__ nodes,
                            const float* __restrict__ syms,
                            const int* __restrict__ en2, const int* __restrict__ es,
                            const float* __restrict__ esg, int din,
                            const _Float16* __restrict__ Wcat,
                            const float* __restrict__ bst,
                            float* __restrict__ Y, int rows) {
  __shared__ _Float16 WS[2][128 * 32];
  const int lane = threadIdx.x & 31;
  const int hseg = lane >> 4;
  const int nlo  = lane & 15;
  const int r0   = blockIdx.x * 128 + (threadIdx.x >> 5) * 16;
  int m = r0 + nlo; if (m >= rows) m = rows - 1;
  int i0 = en2[2 * m], i1 = en2[2 * m + 1], si = es[m];
  float s0 = 1.f, s1 = 1.f, s2 = esg[m];
  if (i0 < 0) { i0 = 0; s0 = 0.f; }
  if (i1 < 0) { i1 = 0; s1 = 0.f; }
  const float* P0 = nodes + (size_t)i0 * 128;
  const float* P1 = nodes + (size_t)i1 * 128;
  const float* P2 = syms  + (size_t)si * 128;
  auto fetch = [&](int k0) {
    const float* p; float sc;
    if (k0 < din)          { p = P0 + k0;            sc = s0; }
    else if (k0 < 2 * din) { p = P1 + (k0 - din);    sc = s1; }
    else                   { p = P2 + (k0 - 2*din);  sc = s2; }
    v8f lo = *(const v8f*)(p + hseg * 8);
    v8f hi = *(const v8f*)(p + 16 + hseg * 8);
    v16h a;
#pragma unroll
    for (int i = 0; i < 8; ++i) { a[i] = (_Float16)(lo[i]*sc); a[8+i] = (_Float16)(hi[i]*sc); }
    return a;
  };
  gemm_core(fetch, Wcat, WS[0], WS[1], 3 * din, Y, rows, r0, bst, nullptr, 0, 1);
}

// ---- tri GEMM (mts123): A row m = [nodes[j0]|nodes[j1]|nodes[j2]], -1 -> 0 ---
__global__ void k_gemm_tri(const float* __restrict__ nodes,
                           const int* __restrict__ sn3, int din,
                           const _Float16* __restrict__ W16,
                           const float* __restrict__ bias,
                           const float* __restrict__ rsgn,
                           float* __restrict__ Y, int rows) {
  __shared__ _Float16 WS[2][128 * 32];
  const int lane = threadIdx.x & 31;
  const int hseg = lane >> 4;
  const int nlo  = lane & 15;
  const int r0   = blockIdx.x * 128 + (threadIdx.x >> 5) * 16;
  int m = r0 + nlo; if (m >= rows) m = rows - 1;
  int j[3]; float s[3];
#pragma unroll
  for (int t = 0; t < 3; ++t) {
    int v = sn3[3 * m + t];
    s[t] = (v < 0) ? 0.f : 1.f;
    j[t] = (v < 0) ? 0 : v;
  }
  const float* P0 = nodes + (size_t)j[0] * 128;
  const float* P1 = nodes + (size_t)j[1] * 128;
  const float* P2 = nodes + (size_t)j[2] * 128;
  auto fetch = [&](int k0) {
    const float* p; float sc;
    if (k0 < din)          { p = P0 + k0;           sc = s[0]; }
    else if (k0 < 2 * din) { p = P1 + (k0 - din);   sc = s[1]; }
    else                   { p = P2 + (k0 - 2*din); sc = s[2]; }
    v8f lo = *(const v8f*)(p + hseg * 8);
    v8f hi = *(const v8f*)(p + 16 + hseg * 8);
    v16h a;
#pragma unroll
    for (int i = 0; i < 8; ++i) { a[i] = (_Float16)(lo[i]*sc); a[8+i] = (_Float16)(hi[i]*sc); }
    return a;
  };
  gemm_core(fetch, W16, WS[0], WS[1], 3 * din, Y, rows, r0, bias, rsgn, 0, 0);
}

// ---------------- elementwise helpers ----------------------------------------
__global__ void k_cvt(const float* __restrict__ s, _Float16* __restrict__ d, int n) {
  int t = blockIdx.x * blockDim.x + threadIdx.x;
  if (t < n) d[t] = (_Float16)s[t];
}
// dst[o][0:Ka]=a[o], dst[o][Ka:Ka+Kb]=b[o]  (f16 concat of two [dout,*] weights)
__global__ void k_cvt_cat(const float* __restrict__ a, int Ka,
                          const float* __restrict__ b, int Kb,
                          _Float16* __restrict__ d, int dout) {
  const int K = Ka + Kb;
  int t = blockIdx.x * blockDim.x + threadIdx.x;
  if (t >= dout * K) return;
  int o = t / K, k = t - o * K;
  float v = (k < Ka) ? a[(size_t)o * Ka + k] : b[(size_t)o * Kb + (k - Ka)];
  d[t] = (_Float16)v;
}
__global__ void k_embed(const float* __restrict__ emb, const int* __restrict__ idx,
                        int dim, float* __restrict__ dst, int rows) {
  int t = blockIdx.x * blockDim.x + threadIdx.x;
  if (t >= rows * dim) return;
  int r = t / dim, c = t - r * dim;
  dst[(size_t)r * 128 + c] = emb[(size_t)idx[r] * dim + c];
}

// ---------------- segmented reductions ----------------------------------------
__global__ void k_seg_clear(float* __restrict__ sum, unsigned* __restrict__ mx,
                            unsigned* __restrict__ mn, float* __restrict__ cnt,
                            int n, int dim) {
  int t = blockIdx.x * blockDim.x + threadIdx.x;
  if (t < n * dim) { sum[t] = 0.f; mx[t] = 0u; mn[t] = 0xFFFFFFFFu; }
  if (t < n) cnt[t] = 0.f;
}
__global__ void k_seg_scatter(const float* __restrict__ src, int srcld,
                              const int* __restrict__ gidx,
                              const int* __restrict__ seg, int rows, int dim,
                              float* __restrict__ sum, unsigned* __restrict__ mx,
                              unsigned* __restrict__ mn, float* __restrict__ cnt) {
  int t = blockIdx.x * blockDim.x + threadIdx.x;
  if (t >= rows * dim) return;
  int row = t / dim, col = t - row * dim;
  int s  = seg[row];
  int sr = gidx ? gidx[row] : row;
  float v = src[(size_t)sr * srcld + col];
  atomicAdd(&sum[(size_t)s * dim + col], v);
  atomicMax(&mx[(size_t)s * dim + col], fenc(v));
  atomicMin(&mn[(size_t)s * dim + col], fenc(v));
  if (col == 0) atomicAdd(&cnt[s], 1.f);
}
// mode 0: [max | mean]; mode 1: [mean | max+min]; mode 2: [max|mean|mean|max+min]
__global__ void k_seg_final(const float* __restrict__ sum, const unsigned* __restrict__ mx,
                            const unsigned* __restrict__ mn, const float* __restrict__ cnt,
                            int n, int dim, int mode, float* __restrict__ out, int outld) {
  int t = blockIdx.x * blockDim.x + threadIdx.x;
  if (t >= n * dim) return;
  int s = t / dim, c = t - s * dim;
  float cn   = cnt[s];
  int   has  = cn > 0.f;
  float mean = sum[(size_t)s * dim + c] / fmaxf(cn, 1.f);
  float M  = has ? fdec(mx[(size_t)s * dim + c]) : 0.f;
  float m_ = has ? fdec(mn[(size_t)s * dim + c]) : 0.f;
  float* o = out + (size_t)s * outld;
  if (mode == 0)      { o[c] = M;    o[dim + c] = mean; }
  else if (mode == 1) { o[c] = mean; o[dim + c] = M + m_; }
  else { o[c] = M; o[dim + c] = mean; o[2*dim + c] = mean; o[3*dim + c] = M + m_; }
}

// ---------------- act + residual + layernorm (D=128, wave per row) ------------
__global__ void k_act_res_ln(const float* __restrict__ acc, const float* __restrict__ res,
                             float* __restrict__ out, int rows, int act) {
  const int row  = blockIdx.x * 8 + (threadIdx.x >> 5);
  const int lane = threadIdx.x & 31;
  if (row >= rows) return;
  const float* a = acc + (size_t)row * 128;
  float v[4];
#pragma unroll
  for (int j = 0; j < 4; ++j) {
    float x = a[lane + 32 * j];
    if (act == 1) x = fmaxf(x, 0.f);
    else if (act == 2) x = tanhf(x);
    if (res) x += res[(size_t)row * 128 + lane + 32 * j];
    v[j] = x;
  }
  float s = v[0] + v[1] + v[2] + v[3];
#pragma unroll
  for (int o = 16; o > 0; o >>= 1) s += __shfl_xor(s, o, 32);
  const float mean = s * (1.f / 128.f);
  float q = 0.f;
#pragma unroll
  for (int j = 0; j < 4; ++j) { float d = v[j] - mean; q += d * d; }
#pragma unroll
  for (int o = 16; o > 0; o >>= 1) q += __shfl_xor(q, o, 32);
  const float rstd = rsqrtf(q * (1.f / 128.f) + 1e-5f);
#pragma unroll
  for (int j = 0; j < 4; ++j)
    out[(size_t)row * 128 + lane + 32 * j] = (v[j] - mean) * rstd;
}

// ---------------- final 128 -> 1 projection (wave per row) --------------------
__global__ void k_dec2(const float* __restrict__ h, const float* __restrict__ w2,
                       const float* __restrict__ b2, float* __restrict__ out, int rows) {
  const int row  = blockIdx.x * 8 + (threadIdx.x >> 5);
  const int lane = threadIdx.x & 31;
  if (row >= rows) return;
  float s = 0.f;
#pragma unroll
  for (int j = 0; j < 4; ++j)
    s += h[(size_t)row * 128 + lane + 32 * j] * w2[lane + 32 * j];
#pragma unroll
  for (int o = 16; o > 0; o >>= 1) s += __shfl_xor(s, o, 32);
  if (lane == 0) out[row] = s + b2[0];
}

// =============================================================================
extern "C" void kernel_launch(void* const* d_in, const int* in_sizes, int n_in,
                              void* d_out, int out_size, void* d_ws, size_t ws_size,
                              hipStream_t stream) {
  (void)in_sizes; (void)n_in; (void)out_size; (void)ws_size;
  int pi = 0;
  auto F = [&]() { return (const float*)d_in[pi++]; };
  auto I = [&]() { return (const int*)d_in[pi++]; };

  // ---- params (pytree leaves, insertion order) ----
  const float* node_emb   = F();
  const float* symbol_emb = F();
  const float* clause_emb = F();
  const float* mc_w[2]      = {F(), F()};
  const float* mc_b[2]      = {F(), F()};
  const float* mct_w[2]     = {F(), F()};
  const float* mts123_w[2]  = {F(), F()};
  const float* mts123_b[2]  = {F(), F()};
  const float* ms_w[2]      = {F(), F()};
  const float* mtsbold_w[2] = {F(), F()};
  const float* bst[2]       = {F(), F()};
  const float* np_w[2][3]; for (int L = 0; L < 2; ++L) for (int e = 0; e < 3; ++e) np_w[L][e] = F();
  const float* sp_w[2][3]; for (int L = 0; L < 2; ++L) for (int e = 0; e < 3; ++e) sp_w[L][e] = F();
  const float* ya_w[2][3]; for (int L = 0; L < 2; ++L) for (int e = 0; e < 3; ++e) ya_w[L][e] = F();
  const float* mtc_w[2] = {F(), F()};
  const float* mt_w[2]  = {F(), F()};
  const float* mt_b[2]  = {F(), F()};
  const float* dec_w1 = F(); const float* dec_b1 = F();
  const float* dec_w2 = F(); const float* dec_b2 = F();

  // ---- graph arrays ----
  const float* edge_sgn = F();     // [3,M]
  const float* sy_sgn   = F();     // [MS]
  const int* ini_nodes   = I();
  const int* ini_symbols = I();
  const int* ini_clauses = I();
  const int* edge_nodes  = I();    // [3,M,2]
  const int* edge_syms   = I();    // [3,M]
  const int* edge_seg    = I();    // [3,M]
  const int* sy_nodes    = I();    // [MS,3]
  const int* sy_seg      = I();
  const int* nc_data = I(); const int* nc_seg = I();
  const int* cl_data = I(); const int* cl_seg = I();
  const int* graph_seg = I();
  (void)I(); // num_graphs (compile-time kB)

  // ---- workspace carve ----
  float* wsf = (float*)d_ws;
  size_t off = 0;
  auto alloc = [&](size_t n) { float* p = wsf + off; off += n; return p; };
  float* nodesA = alloc((size_t)kN * 128);
  float* nodesB = alloc((size_t)kN * 128);
  float* symsA  = alloc((size_t)kS * 128);
  float* symsB  = alloc((size_t)kS * 128);
  float* clsA   = alloc((size_t)kC * 128);
  float* clsB   = alloc((size_t)kC * 128);
  float* accN   = alloc((size_t)kN * 128);
  float* accS   = alloc((size_t)kS * 128);
  float* accC   = alloc((size_t)kC * 128);
  float* zbuf   = alloc((size_t)kM * 128);
  float* xsbuf  = alloc((size_t)kMS * 128);
  float* segsum = alloc((size_t)kN * 128);
  unsigned* segmax = (unsigned*)alloc((size_t)kN * 128);
  unsigned* segmin = (unsigned*)alloc((size_t)kN * 128);
  float* segcnt = alloc((size_t)kN);
  float* segout = alloc((size_t)kN * 256);
  float* pooled = alloc((size_t)kB * 512);
  float* hdec   = alloc((size_t)kB * 128);
  _Float16* w16 = (_Float16*)alloc((size_t)128 * 512 / 2 + 64);

  auto g1 = [](long long n) { return dim3((unsigned)((n + 255) / 256)); };
  const dim3 B256(256);
  const dim3 gN((kN + 127) / 128), gM((kM + 127) / 128), gS((kS + 127) / 128),
             gC((kC + 127) / 128), gMS((kMS + 127) / 128);

  // ---- embeddings ----
  k_embed<<<g1((long long)kN * 64), B256, 0, stream>>>(node_emb,   ini_nodes,   64, nodesA, kN);
  k_embed<<<g1((long long)kS * 64), B256, 0, stream>>>(symbol_emb, ini_symbols, 64, symsA,  kS);
  k_embed<<<g1((long long)kC * 64), B256, 0, stream>>>(clause_emb, ini_clauses, 64, clsA,   kC);

  float *ncur = nodesA, *nnext = nodesB, *scur = symsA, *snext = symsB,
        *ccur = clsA, *cnext = clsB;

  for (int L = 0; L < 2; ++L) {
    const int din = L ? 128 : 64;
    const int K2 = 2 * din, K3 = 3 * din;

    // acc = nodes @ mt^T + mt_b
    k_cvt<<<g1(128 * din), B256, 0, stream>>>(mt_w[L], w16, 128 * din);
    k_gemm_plain<<<gN, B256, 0, stream>>>(ncur, 128, w16, din, mt_b[L], nullptr, accN, kN, 0, 0);

    // edge messages + segment max/mean + ya projection
    for (int e = 0; e < 3; ++e) {
      k_cvt_cat<<<g1(128 * K3), B256, 0, stream>>>(np_w[L][e], K2, sp_w[L][e], din, w16, 128);
      k_gemm_edge<<<gM, B256, 0, stream>>>(ncur, scur,
          edge_nodes + (size_t)e * kM * 2, edge_syms + (size_t)e * kM,
          edge_sgn + (size_t)e * kM, din, w16, bst[L], zbuf, kM);
      k_seg_clear<<<g1((long long)kN * 128), B256, 0, stream>>>(segsum, segmax, segmin, segcnt, kN, 128);
      k_seg_scatter<<<g1((long long)kM * 128), B256, 0, stream>>>(zbuf, 128, nullptr,
          edge_seg + (size_t)e * kM, kM, 128, segsum, segmax, segmin, segcnt);
      k_seg_final<<<g1((long long)kN * 128), B256, 0, stream>>>(segsum, segmax, segmin, segcnt,
          kN, 128, 0, segout, 256);
      k_cvt<<<g1(128 * 256), B256, 0, stream>>>(ya_w[L][e], w16, 128 * 256);
      k_gemm_plain<<<gN, B256, 0, stream>>>(segout, 256, w16, 256, nullptr, nullptr, accN, kN, 1, 0);
    }

    // clause -> node aggregation (mtc)
    k_seg_clear<<<g1((long long)kN * din), B256, 0, stream>>>(segsum, segmax, segmin, segcnt, kN, din);
    k_seg_scatter<<<g1((long long)kNC * din), B256, 0, stream>>>(ccur, 128, nc_data, nc_seg,
        kNC, din, segsum, segmax, segmin, segcnt);
    k_seg_final<<<g1((long long)kN * din), B256, 0, stream>>>(segsum, segmax, segmin, segcnt,
        kN, din, 0, segout, K2);
    k_cvt<<<g1(128 * K2), B256, 0, stream>>>(mtc_w[L], w16, 128 * K2);
    k_gemm_plain<<<gN, B256, 0, stream>>>(segout, K2, w16, K2, nullptr, nullptr, accN, kN, 1, 0);

    // symbols: mts123 -> seg_prime -> ms + mtsbold
    k_cvt<<<g1(128 * K3), B256, 0, stream>>>(mts123_w[L], w16, 128 * K3);
    k_gemm_tri<<<gMS, B256, 0, stream>>>(ncur, sy_nodes, din, w16, mts123_b[L], sy_sgn, xsbuf, kMS);
    k_seg_clear<<<g1((long long)kS * 128), B256, 0, stream>>>(segsum, segmax, segmin, segcnt, kS, 128);
    k_seg_scatter<<<g1((long long)kMS * 128), B256, 0, stream>>>(xsbuf, 128, nullptr, sy_seg,
        kMS, 128, segsum, segmax, segmin, segcnt);
    k_seg_final<<<g1((long long)kS * 128), B256, 0, stream>>>(segsum, segmax, segmin, segcnt,
        kS, 128, 1, segout, 256);
    k_cvt<<<g1(128 * din), B256, 0, stream>>>(ms_w[L], w16, 128 * din);
    k_gemm_plain<<<gS, B256, 0, stream>>>(scur, 128, w16, din, nullptr, nullptr, accS, kS, 0, 0);
    k_cvt<<<g1(128 * 256), B256, 0, stream>>>(mtsbold_w[L], w16, 128 * 256);
    k_gemm_plain<<<gS, B256, 0, stream>>>(segout, 256, w16, 256, nullptr, nullptr, accS, kS, 1, 0);

    // clauses: node aggregation + mc
    k_seg_clear<<<g1((long long)kC * din), B256, 0, stream>>>(segsum, segmax, segmin, segcnt, kC, din);
    k_seg_scatter<<<g1((long long)kCN * din), B256, 0, stream>>>(ncur, 128, cl_data, cl_seg,
        kCN, din, segsum, segmax, segmin, segcnt);
    k_seg_final<<<g1((long long)kC * din), B256, 0, stream>>>(segsum, segmax, segmin, segcnt,
        kC, din, 0, segout, K2);
    k_cvt<<<g1(128 * din), B256, 0, stream>>>(mc_w[L], w16, 128 * din);
    k_gemm_plain<<<gC, B256, 0, stream>>>(ccur, 128, w16, din, mc_b[L], nullptr, accC, kC, 0, 0);
    k_cvt<<<g1(128 * K2), B256, 0, stream>>>(mct_w[L], w16, 128 * K2);
    k_gemm_plain<<<gC, B256, 0, stream>>>(segout, K2, w16, K2, nullptr, nullptr, accC, kC, 1, 0);

    // activation + residual + layernorm
    k_act_res_ln<<<dim3((kN + 7) / 8), B256, 0, stream>>>(accN, L ? ncur : nullptr, nnext, kN, 1);
    k_act_res_ln<<<dim3((kS + 7) / 8), B256, 0, stream>>>(accS, L ? scur : nullptr, snext, kS, 2);
    k_act_res_ln<<<dim3((kC + 7) / 8), B256, 0, stream>>>(accC, L ? ccur : nullptr, cnext, kC, 1);
    { float* t = ncur; ncur = nnext; nnext = t; }
    { float* t = scur; scur = snext; snext = t; }
    { float* t = ccur; ccur = cnext; cnext = t; }
  }

  // ---- graph pooling: [max|mean|mean|max+min] over clauses ----
  k_seg_clear<<<g1((long long)kB * 128), B256, 0, stream>>>(segsum, segmax, segmin, segcnt, kB, 128);
  k_seg_scatter<<<g1((long long)kC * 128), B256, 0, stream>>>(ccur, 128, nullptr, graph_seg,
      kC, 128, segsum, segmax, segmin, segcnt);
  k_seg_final<<<g1((long long)kB * 128), B256, 0, stream>>>(segsum, segmax, segmin, segcnt,
      kB, 128, 2, pooled, 512);

  // ---- decode ----
  k_cvt<<<g1(128 * 512), B256, 0, stream>>>(dec_w1, w16, 128 * 512);
  k_gemm_plain<<<dim3(1), B256, 0, stream>>>(pooled, 512, w16, 512, dec_b1, nullptr, hdec, kB, 0, 1);
  k_dec2<<<dim3((kB + 7) / 8), B256, 0, stream>>>(hdec, dec_w2, dec_b2, (float*)d_out, kB);
}
